// GAT_6227702579509
// MI455X (gfx1250) — matrix operand order repaired and checked by reading, not compiled, not run on hardware
//
#include <hip/hip_runtime.h>
#include <math.h>

typedef __attribute__((ext_vector_type(16))) _Float16 v16h;
typedef __attribute__((ext_vector_type(8)))  _Float16 v8h;
typedef __attribute__((ext_vector_type(8)))  float    v8f;

#define N_NODES 50000
#define N_EDGES 800000
#define H       128
#define H3      384
#define H4      512
#define SCALE_F 30.0f

// ---------- helpers ----------

__device__ __forceinline__ float gelu_f(float x) {
    // exact erf GELU (torch default)
    return 0.5f * x * (1.0f + erff(x * 0.70710678118654752440f));
}

// A-fragment (16x32 f16, MxK) in this code's contiguous k order, same as its B loads:
// lane<16 holds row M=lane, K = {0..15}; lane>=16 same row M=lane-16, K = {16..31}
__device__ __forceinline__ v16h load_a16(const _Float16* rowk /* &x[row][ktile*32] */, int lhi) {
    v8h lo = *(const v8h*)(rowk + lhi * 16);
    v8h hi = *(const v8h*)(rowk + lhi * 16 + 8);
    return __builtin_shufflevector(lo, hi, 0,1,2,3,4,5,6,7,8,9,10,11,12,13,14,15);
}

__device__ __forceinline__ v8f wmma_f16(v16h a, v16h b, v8f c) {
    return __builtin_amdgcn_wmma_f32_16x16x32_f16(false, a, false, b, (short)0, c, false, false);
}

// ---------- zero scratch ----------

__global__ void zero_kernel(float* __restrict__ p, long long n) {
    long long i = (long long)blockIdx.x * blockDim.x + threadIdx.x;
    if (i < n) p[i] = 0.0f;
}

// ---------- edge kernel: gather-concat -> 3-layer MLP (WMMA f16) -> scaled scatter-add ----------

// LDS layout (halves, padded strides to dodge 64-bank conflicts)
#define W1S (H3 + 8)   // 392
#define WSS (H  + 8)   // 136
#define EDGE_LDS_BYTES 194048

__global__ void __launch_bounds__(256)
edge_kernel(const float* __restrict__ hV, const float* __restrict__ hE,
            const int* __restrict__ srcIdx, const int* __restrict__ dstIdx,
            const float* __restrict__ W1, const float* __restrict__ b1,
            const float* __restrict__ W2, const float* __restrict__ b2,
            const float* __restrict__ W3, const float* __restrict__ b3,
            const float* __restrict__ Aw,
            float* __restrict__ dhacc, float* __restrict__ esum)
{
    extern __shared__ char smem[];
    _Float16* w1 = (_Float16*)smem;            // 128 x W1S
    _Float16* w2 = w1 + 128 * W1S;             // 128 x WSS
    _Float16* w3 = w2 + 128 * WSS;             // 128 x WSS
    _Float16* xt = w3 + 128 * WSS;             // 16 x W1S   (concat input tile)
    _Float16* x1 = xt + 16  * W1S;             // 16 x WSS
    _Float16* x2 = x1 + 16  * WSS;             // 16 x WSS
    float* aLds  = (float*)(x2 + 16 * WSS);    // 384
    float* part  = aLds + H3;                  // 256
    float* fE    = part + 256;                 // 16
    int*   sSrc  = (int*)(fE + 16);            // 16
    int*   sDst  = sSrc + 16;                  // 16

    const int tid  = threadIdx.x;
    const int lane = tid & 31;
    const int wave = tid >> 5;
    const int lrow = lane & 15;
    const int lhi  = lane >> 4;

    // stage weights once per WG: f32 global -> f16 padded LDS
    for (int i = tid; i < 128 * H3; i += 256)
        w1[(i / H3) * W1S + (i % H3)] = (_Float16)W1[i];
    for (int i = tid; i < 128 * H; i += 256) {
        int r = i >> 7, c = i & (H - 1);
        w2[r * WSS + c] = (_Float16)W2[i];
        w3[r * WSS + c] = (_Float16)W3[i];
    }
    for (int i = tid; i < H3; i += 256) aLds[i] = Aw[i];
    __syncthreads();

    for (int tile = blockIdx.x; tile < N_EDGES / 16; tile += gridDim.x) {
        const int base = tile * 16;
        if (tid < 16) { sSrc[tid] = srcIdx[base + tid]; sDst[tid] = dstIdx[base + tid]; }
        __syncthreads();

        // gather/concat -> f16 LDS tile; f32 partial dot with A
        {
            const int row = tid >> 4, seg = tid & 15;
            const long long e = base + row;
            float acc = 0.0f;
            #pragma unroll
            for (int j = 0; j < 24; ++j) {
                const int c = seg * 24 + j;
                float x;
                if (c < H)            x = hV[(long long)sSrc[row] * H + c];
                else if (c < 2 * H)   x = hE[e * H + (c - H)];
                else                  x = hV[(long long)sDst[row] * H + (c - 2 * H)];
                xt[row * W1S + c] = (_Float16)x;
                acc += x * aLds[c];
            }
            part[tid] = acc;
        }
        __syncthreads();
        if (tid < 16) {
            float s = 0.0f;
            #pragma unroll
            for (int j = 0; j < 16; ++j) s += part[tid * 16 + j];
            s = (s > 0.0f) ? s : 0.01f * s;          // leaky_relu
            s = 1.0f / (1.0f + expf(-s));            // sigmoid
            fE[tid] = expf(s);                       // unnormalized e
        }
        __syncthreads();
        if (tid == 0) {
            float se = 0.0f;
            #pragma unroll
            for (int r = 0; r < 16; ++r) se += fE[r];
            atomicAdd(esum, se);
        }

        // layer 1: [16x384] x [384x128], wave owns 16-feature n-tile
        {
            v8f acc = {0.f,0.f,0.f,0.f,0.f,0.f,0.f,0.f};
            #pragma unroll
            for (int kt = 0; kt < 12; ++kt) {
                v16h a = load_a16(xt + lrow * W1S + kt * 32, lhi);
                v16h b = *(const v16h*)(w1 + (wave * 16 + lrow) * W1S + kt * 32 + lhi * 16);
                acc = wmma_f16(a, b, acc);
            }
            const int n = wave * 16 + lrow;
            const float bias = b1[n];
            #pragma unroll
            for (int r = 0; r < 8; ++r)
                x1[(r + 8 * lhi) * WSS + n] = (_Float16)gelu_f(acc[r] + bias);
        }
        __syncthreads();

        // layer 2: [16x128] x [128x128]
        {
            v8f acc = {0.f,0.f,0.f,0.f,0.f,0.f,0.f,0.f};
            #pragma unroll
            for (int kt = 0; kt < 4; ++kt) {
                v16h a = load_a16(x1 + lrow * WSS + kt * 32, lhi);
                v16h b = *(const v16h*)(w2 + (wave * 16 + lrow) * WSS + kt * 32 + lhi * 16);
                acc = wmma_f16(a, b, acc);
            }
            const int n = wave * 16 + lrow;
            const float bias = b2[n];
            #pragma unroll
            for (int r = 0; r < 8; ++r)
                x2[(r + 8 * lhi) * WSS + n] = (_Float16)gelu_f(acc[r] + bias);
        }
        __syncthreads();

        // layer 3 + scaled scatter-add to source nodes
        {
            v8f acc = {0.f,0.f,0.f,0.f,0.f,0.f,0.f,0.f};
            #pragma unroll
            for (int kt = 0; kt < 4; ++kt) {
                v16h a = load_a16(x2 + lrow * WSS + kt * 32, lhi);
                v16h b = *(const v16h*)(w3 + (wave * 16 + lrow) * WSS + kt * 32 + lhi * 16);
                acc = wmma_f16(a, b, acc);
            }
            const int n = wave * 16 + lrow;
            const float bias = b3[n];
            #pragma unroll
            for (int r = 0; r < 8; ++r) {
                const int m = r + 8 * lhi;
                const float val = (acc[r] + bias) * fE[m];
                atomicAdd(&dhacc[(long long)sSrc[m] * H + n], val);
            }
        }
        __syncthreads();
    }
}

// ---------- node kernel: dh normalize + LN1 + FFN (WMMA) + LN2 ----------

#define WSO (H4 + 8)   // 520
#define VFS 132
#define NODE_LDS_BYTES 304000

__global__ void __launch_bounds__(256)
node_kernel(const float* __restrict__ hV,
            const float* __restrict__ Win,  const float* __restrict__ binb,
            const float* __restrict__ Wout, const float* __restrict__ boutb,
            const float* __restrict__ ln1w, const float* __restrict__ ln1b,
            const float* __restrict__ ln2w, const float* __restrict__ ln2b,
            const float* __restrict__ dhacc, const float* __restrict__ esum,
            float* __restrict__ out)
{
    extern __shared__ char smem[];
    _Float16* win  = (_Float16*)smem;           // 512 x WSS
    _Float16* wout = win + H4 * WSS;            // 128 x WSO
    _Float16* v1h  = wout + H * WSO;            // 16 x WSS
    _Float16* tmid = v1h + 16 * WSS;            // 16 x WSO
    float* v1f = (float*)(tmid + 16 * WSO);     // 16 x VFS (f32 residual / y)
    float* pA  = v1f + 16 * VFS;                // 256
    float* pB  = pA + 256;                      // 256
    float* muA = pB + 256;                      // 16
    float* rsA = muA + 16;                      // 16

    const int tid  = threadIdx.x;
    const int lane = tid & 31;
    const int wave = tid >> 5;
    const int lrow = lane & 15;
    const int lhi  = lane >> 4;

    for (int i = tid; i < H4 * H; i += 256) {
        { int r = i >> 7, c = i & (H - 1);  win[r * WSS + c]  = (_Float16)Win[i];  }  // (512,128)
        { int r = i >> 9, c = i & (H4 - 1); wout[r * WSO + c] = (_Float16)Wout[i]; }  // (128,512)
    }
    __syncthreads();
    const float invE = 1.0f / (esum[0] * SCALE_F);

    for (int tile = blockIdx.x; tile < N_NODES / 16; tile += gridDim.x) {
        const int base = tile * 16;
        const int row = tid >> 4, seg = tid & 15;

        // residual + LN1 (f32, two-phase reduction over 128 features)
        {
            const long long node = base + row;
            float v[8]; float s = 0.0f, q = 0.0f;
            #pragma unroll
            for (int j = 0; j < 8; ++j) {
                const int c = seg * 8 + j;
                float x = hV[node * H + c] + dhacc[node * H + c] * invE;
                v[j] = x; s += x; q += x * x;
            }
            pA[tid] = s; pB[tid] = q;
            __syncthreads();
            if (tid < 16) {
                float m = 0.0f, qq = 0.0f;
                #pragma unroll
                for (int j = 0; j < 16; ++j) { m += pA[tid * 16 + j]; qq += pB[tid * 16 + j]; }
                m *= (1.0f / H); qq = qq * (1.0f / H) - m * m;
                muA[tid] = m; rsA[tid] = rsqrtf(qq + 1e-5f);
            }
            __syncthreads();
            const float mu = muA[row], rs = rsA[row];
            #pragma unroll
            for (int j = 0; j < 8; ++j) {
                const int c = seg * 8 + j;
                float y = (v[j] - mu) * rs * ln1w[c] + ln1b[c];
                v1f[row * VFS + c] = y;
                v1h[row * WSS + c] = (_Float16)y;
            }
        }
        __syncthreads();

        // FFN in: [16x128] x [128x512]; wave owns 4 n-tiles of 16
        #pragma unroll
        for (int i = 0; i < 4; ++i) {
            const int nt = wave * 4 + i;
            v8f acc = {0.f,0.f,0.f,0.f,0.f,0.f,0.f,0.f};
            #pragma unroll
            for (int kt = 0; kt < 4; ++kt) {
                v16h a = load_a16(v1h + lrow * WSS + kt * 32, lhi);
                v16h b = *(const v16h*)(win + (nt * 16 + lrow) * WSS + kt * 32 + lhi * 16);
                acc = wmma_f16(a, b, acc);
            }
            const int n = nt * 16 + lrow;
            const float bias = binb[n];
            #pragma unroll
            for (int r = 0; r < 8; ++r)
                tmid[(r + 8 * lhi) * WSO + n] = (_Float16)gelu_f(acc[r] + bias);
        }
        __syncthreads();

        // FFN out: [16x512] x [512x128]; wave owns one n-tile; add residual into v1f
        {
            v8f acc = {0.f,0.f,0.f,0.f,0.f,0.f,0.f,0.f};
            #pragma unroll
            for (int kt = 0; kt < 16; ++kt) {
                v16h a = load_a16(tmid + lrow * WSO + kt * 32, lhi);
                v16h b = *(const v16h*)(wout + (wave * 16 + lrow) * WSO + kt * 32 + lhi * 16);
                acc = wmma_f16(a, b, acc);
            }
            const int n = wave * 16 + lrow;
            const float bias = boutb[n];
            #pragma unroll
            for (int r = 0; r < 8; ++r) {
                const int m = r + 8 * lhi;
                v1f[m * VFS + n] += acc[r] + bias;   // y = v1 + ffn
            }
        }
        __syncthreads();

        // LN2 + write out
        {
            float v[8]; float s = 0.0f, q = 0.0f;
            #pragma unroll
            for (int j = 0; j < 8; ++j) {
                float x = v1f[row * VFS + seg * 8 + j];
                v[j] = x; s += x; q += x * x;
            }
            pA[tid] = s; pB[tid] = q;
            __syncthreads();
            if (tid < 16) {
                float m = 0.0f, qq = 0.0f;
                #pragma unroll
                for (int j = 0; j < 16; ++j) { m += pA[tid * 16 + j]; qq += pB[tid * 16 + j]; }
                m *= (1.0f / H); qq = qq * (1.0f / H) - m * m;
                muA[tid] = m; rsA[tid] = rsqrtf(qq + 1e-5f);
            }
            __syncthreads();
            const float mu = muA[row], rs = rsA[row];
            const long long node = base + row;
            #pragma unroll
            for (int j = 0; j < 8; ++j) {
                const int c = seg * 8 + j;
                out[node * H + c] = (v[j] - mu) * rs * ln2w[c] + ln2b[c];
            }
        }
        __syncthreads();
    }
}

// ---------- launcher ----------

extern "C" void kernel_launch(void* const* d_in, const int* in_sizes, int n_in,
                              void* d_out, int out_size, void* d_ws, size_t ws_size,
                              hipStream_t stream)
{
    (void)in_sizes; (void)n_in; (void)out_size; (void)ws_size;
    const float* hV    = (const float*)d_in[0];
    const float* hE    = (const float*)d_in[1];
    const int*   srcI  = (const int*)d_in[2];
    const int*   dstI  = (const int*)d_in[3];
    /* d_in[4] batch_id unused by reference math */
    const float* W1    = (const float*)d_in[5];
    const float* b1    = (const float*)d_in[6];
    const float* W2    = (const float*)d_in[7];
    const float* b2    = (const float*)d_in[8];
    const float* W3    = (const float*)d_in[9];
    const float* b3    = (const float*)d_in[10];
    const float* Aw    = (const float*)d_in[11];
    const float* ln1w  = (const float*)d_in[12];
    const float* ln1b  = (const float*)d_in[13];
    const float* ln2w  = (const float*)d_in[14];
    const float* ln2b  = (const float*)d_in[15];
    const float* Win   = (const float*)d_in[16];
    const float* binb  = (const float*)d_in[17];
    const float* Wout  = (const float*)d_in[18];
    const float* boutb = (const float*)d_in[19];

    float* dhacc = (float*)d_ws;                       // N*H f32 accumulator
    float* esum  = dhacc + (long long)N_NODES * H;     // 1 f32

    const long long ztotal = (long long)N_NODES * H + 1;
    zero_kernel<<<(int)((ztotal + 255) / 256), 256, 0, stream>>>(dhacc, ztotal);

    edge_kernel<<<1024, 256, EDGE_LDS_BYTES, stream>>>(
        hV, hE, srcI, dstI, W1, b1, W2, b2, W3, b3, Aw, dhacc, esum);

    node_kernel<<<512, 256, NODE_LDS_BYTES, stream>>>(
        hV, Win, binb, Wout, boutb, ln1w, ln1b, ln2w, ln2b, dhacc, esum, (float*)d_out);
}
